// MultiDconvHeadTransposedAttention_59072980189966
// MI455X (gfx1250) — matrix operand order repaired and checked by reading, not compiled, not run on hardware
//
#include <hip/hip_runtime.h>

// ---------------------------------------------------------------------------
// MDTA (transposed attention) for MI455X / gfx1250, wave32 + WMMA f16->f32.
// Workspace layout (bytes). Requires ws_size >= ~177 MB.
// ---------------------------------------------------------------------------
#define OFF_XH    0ULL           /* xh  f16 [65536][192]  = 25165824 ; reused as OUT2H after GEMM1 */
#define OFF_WQT   25165824ULL    /* w_qkv^T f16 [576][192] = 221184 */
#define OFF_WPT   25387008ULL    /* w_proj^T f16 [192][192] = 73728 */
#define OFF_QKVP  25460736ULL    /* qkv planar f16 [576][65536] = 75497472 */
#define OFF_Q     100958208ULL   /* q planar f16 [16][48][16384] = 25165824 */
#define OFF_K     126124032ULL   /* k planar f16 [16][48][16384] = 25165824 */
#define OFF_V     151289856ULL   /* v planar f16 [16][48][16384] = 25165824 */
#define OFF_RQ    176455680ULL   /* 768 f32 */
#define OFF_RK    176458752ULL   /* 768 f32 */
#define OFF_ATTN  176461824ULL   /* attn f16 [16][48][64] = 98304 */
#define OFF_ATTNF 176560128ULL   /* attn f32 accum [16][48][48] = 147456 ; end = 176707584 */

typedef _Float16 half_t;
typedef _Float16 v8h  __attribute__((ext_vector_type(8)));
typedef _Float16 v16h __attribute__((ext_vector_type(16)));
typedef float    v8f  __attribute__((ext_vector_type(8)));

static __device__ __forceinline__ v16h cat8(v8h lo, v8h hi) {
  return __builtin_shufflevector(lo, hi, 0,1,2,3,4,5,6,7,8,9,10,11,12,13,14,15);
}
static __device__ __forceinline__ v8f wmma_f16f32(v16h a, v16h b, v8f c) {
  // emits v_wmma_f32_16x16x32_f16
  return __builtin_amdgcn_wmma_f32_16x16x32_f16(false, a, false, b, (short)0, c, false, false);
}
static __device__ __forceinline__ v8f vzero8() {
  v8f z = {0.f,0.f,0.f,0.f,0.f,0.f,0.f,0.f};
  return z;
}
// A fragment 16x32 (f16) from row-major [M][lda]: lane m=m0+(L&15),
// halves k = kb..kb+15 with kb = k0 + (L>>4)*16
static __device__ __forceinline__ v16h load_a(const half_t* __restrict__ base,
                                              int lda, int m0, int k0, int lane) {
  int m  = m0 + (lane & 15);
  int kb = k0 + ((lane >> 4) << 4);
  const half_t* p = base + (size_t)m * (size_t)lda + kb;
  v8h lo = *(const v8h*)p;
  v8h hi = *(const v8h*)(p + 8);
  return cat8(lo, hi);
}
// B fragment 32x16 (f16) from K-transposed BT[N][ldb]: lane n=n0+(L&15),
// halves k = kb..kb+15 with kb = k0 + (L>>4)*16
static __device__ __forceinline__ v16h load_bt(const half_t* __restrict__ baseT,
                                               int ldb, int n0, int k0, int lane) {
  int n  = n0 + (lane & 15);
  int kb = k0 + ((lane >> 4) << 4);
  const v8h* p = (const v8h*)(baseT + (size_t)n * (size_t)ldb + kb);
  return cat8(p[0], p[1]);
}

// ------------------------- conversion kernels ------------------------------
__global__ void __launch_bounds__(256) k_cvt_x(const float* __restrict__ x,
                                               half_t* __restrict__ xh, int n4) {
  int i = (blockIdx.x * 256 + threadIdx.x) << 2;
  if (i < n4 * 4) {
    float4 v = *(const float4*)(x + i);
    xh[i + 0] = (half_t)v.x; xh[i + 1] = (half_t)v.y;
    xh[i + 2] = (half_t)v.z; xh[i + 3] = (half_t)v.w;
  }
}
// w[K][N] (f32 row-major) -> wT[N][K] (f16)
__global__ void __launch_bounds__(256) k_transpose_w(const float* __restrict__ w,
                                                     half_t* __restrict__ wT,
                                                     int K, int N) {
  int i = blockIdx.x * 256 + threadIdx.x;
  if (i < K * N) {
    int k = i / N, n = i - k * N;
    wT[(size_t)n * K + k] = (half_t)w[i];
  }
}
__global__ void __launch_bounds__(256) k_zero_f32(float* __restrict__ p, int n) {
  int i = blockIdx.x * 256 + threadIdx.x;
  if (i < n) p[i] = 0.f;
}

// ------------------------- qkv 1x1 conv GEMM -------------------------------
// D[m][n] = sum_k xh[m][k] * w_qkv[k][n];  M=65536, N=576, K=192
// Output stored PLANAR: qkvP[n][m]  (one b128 store per 16x16 D tile per lane)
__global__ void __launch_bounds__(256) k_gemm_qkv(const half_t* __restrict__ xh,
                                                  const half_t* __restrict__ wT,
                                                  half_t* __restrict__ qkvP) {
  const int lane = threadIdx.x & 31, wave = threadIdx.x >> 5;
  const int m0 = (blockIdx.x * 8 + wave) << 4;   // 512 blocks x 8 waves
  const int n0 = blockIdx.y << 6;                // 9 blocks of 64
  v8f acc[4];
  #pragma unroll
  for (int j = 0; j < 4; ++j) acc[j] = vzero8();
  #pragma unroll
  for (int k0 = 0; k0 < 192; k0 += 32) {
    v16h a = load_a(xh, 192, m0, k0, lane);
    #pragma unroll
    for (int j = 0; j < 4; ++j) {
      v16h b = load_bt(wT, 192, n0 + (j << 4), k0, lane);
      acc[j] = wmma_f16f32(a, b, acc[j]);
    }
  }
  const int mrow = m0 + ((lane >> 4) << 3);
  #pragma unroll
  for (int j = 0; j < 4; ++j) {
    int n = n0 + (j << 4) + (lane & 15);
    v8h h;
    #pragma unroll
    for (int r = 0; r < 8; ++r) h[r] = (half_t)acc[j][r];
    *(v8h*)(qkvP + (size_t)n * 65536 + mrow) = h;
  }
}

// ------------------------- depthwise 3x3 SAME ------------------------------
// in: qkvP planar [c=576][b=4][128][128]; out: q/k/v planar [bh][ch][hw]
__global__ void __launch_bounds__(256) k_dwconv(const half_t* __restrict__ qkvP,
                                                const float* __restrict__ wdw,
                                                half_t* __restrict__ qbuf,
                                                half_t* __restrict__ kbuf,
                                                half_t* __restrict__ vbuf) {
  const int c = blockIdx.y;                       // 0..575
  const int m = blockIdx.x * 256 + threadIdx.x;   // 0..65535 = b*16384+hw
  const int hw = m & 16383;
  const int y = hw >> 7, x = hw & 127;
  const half_t* plane = qkvP + (size_t)c * 65536 + (m - hw);
  float acc = 0.f;
  #pragma unroll
  for (int ky = 0; ky < 3; ++ky) {
    int yy = y + ky - 1;
    if ((unsigned)yy >= 128u) continue;
    #pragma unroll
    for (int kx = 0; kx < 3; ++kx) {
      int xx = x + kx - 1;
      if ((unsigned)xx >= 128u) continue;
      acc += (float)plane[yy * 128 + xx] * wdw[(ky * 3 + kx) * 576 + c];
    }
  }
  const int sel = c / 192, cc = c % 192;
  const int head = cc / 48, ch = cc - head * 48;
  const int b = m >> 14;
  const size_t idx = (((size_t)(b * 4 + head) * 48 + ch) << 14) + hw;
  if (sel == 0)      qbuf[idx] = (half_t)acc;
  else if (sel == 1) kbuf[idx] = (half_t)acc;
  else               vbuf[idx] = (half_t)acc;
}

// ------------------------- row L2-norm (rsqrt of sumsq) --------------------
// qbuf and kbuf are contiguous in ws: 1536 rows of 16384 halves.
__global__ void __launch_bounds__(256) k_rownorm(const half_t* __restrict__ rows,
                                                 float* __restrict__ rn) {
  __shared__ float red[256];
  const half_t* p = rows + ((size_t)blockIdx.x << 14);
  float s = 0.f;
  for (int i = threadIdx.x; i < 16384; i += 256) {
    float v = (float)p[i];
    s += v * v;
  }
  red[threadIdx.x] = s;
  __syncthreads();
  for (int off = 128; off > 0; off >>= 1) {
    if (threadIdx.x < off) red[threadIdx.x] += red[threadIdx.x + off];
    __syncthreads();
  }
  if (threadIdx.x == 0) rn[blockIdx.x] = rsqrtf(red[0] + 1e-20f);
}

// ------------------------- attn partial: q k^T -----------------------------
// Grid (16 bh, 8 kchunk): each block reduces a K-range of 2048 into a 48x48
// f32 tile (LDS ds_add_f32), then global_atomic_add_f32 into attnF[bh].
__global__ void __launch_bounds__(256) k_attn_partial(const half_t* __restrict__ qbuf,
                                                      const half_t* __restrict__ kbuf,
                                                      float* __restrict__ attnF) {
  __shared__ float accS[48 * 48];
  const int bh = blockIdx.x;
  const half_t* qp = qbuf + ((size_t)bh * 48 << 14);
  const half_t* kp = kbuf + ((size_t)bh * 48 << 14);
  const int lane = threadIdx.x & 31, wave = threadIdx.x >> 5;

  for (int i = threadIdx.x; i < 48 * 48; i += 256) accS[i] = 0.f;
  __syncthreads();

  v8f acc[3][3];
  #pragma unroll
  for (int i = 0; i < 3; ++i)
    #pragma unroll
    for (int j = 0; j < 3; ++j) acc[i][j] = vzero8();

  const int kbase = blockIdx.y * 2048 + wave * 256;
  #pragma unroll
  for (int it = 0; it < 8; ++it) {
    const int k0 = kbase + it * 32;
    v16h a0 = load_a(qp, 16384,  0, k0, lane);
    v16h a1 = load_a(qp, 16384, 16, k0, lane);
    v16h a2 = load_a(qp, 16384, 32, k0, lane);
    v16h b0 = load_bt(kp, 16384,  0, k0, lane);
    v16h b1 = load_bt(kp, 16384, 16, k0, lane);
    v16h b2 = load_bt(kp, 16384, 32, k0, lane);
    acc[0][0] = wmma_f16f32(a0, b0, acc[0][0]);
    acc[0][1] = wmma_f16f32(a0, b1, acc[0][1]);
    acc[0][2] = wmma_f16f32(a0, b2, acc[0][2]);
    acc[1][0] = wmma_f16f32(a1, b0, acc[1][0]);
    acc[1][1] = wmma_f16f32(a1, b1, acc[1][1]);
    acc[1][2] = wmma_f16f32(a1, b2, acc[1][2]);
    acc[2][0] = wmma_f16f32(a2, b0, acc[2][0]);
    acc[2][1] = wmma_f16f32(a2, b1, acc[2][1]);
    acc[2][2] = wmma_f16f32(a2, b2, acc[2][2]);
  }
  // reduce partial 48x48 tiles across waves (ds_add_f32)
  #pragma unroll
  for (int i = 0; i < 3; ++i)
    #pragma unroll
    for (int j = 0; j < 3; ++j)
      #pragma unroll
      for (int r = 0; r < 8; ++r) {
        int mm = i * 16 + ((lane >> 4) << 3) + r;
        int nn = j * 16 + (lane & 15);
        atomicAdd(&accS[mm * 48 + nn], acc[i][j][r]);
      }
  __syncthreads();
  // block partial -> global accumulator (global_atomic_add_f32)
  float* dst = attnF + (size_t)bh * 48 * 48;
  for (int i = threadIdx.x; i < 48 * 48; i += 256) atomicAdd(&dst[i], accS[i]);
}

// ------------------------- softmax + norm scales ---------------------------
// attn[c][d] = softmax_d( attnF[c][d] * rq[c] * rk[d] * temp[head] ), f16 out
__global__ void __launch_bounds__(64) k_softmax(const float* __restrict__ attnF,
                                                const float* __restrict__ rq,
                                                const float* __restrict__ rk,
                                                const float* __restrict__ temp,
                                                half_t* __restrict__ attnh) {
  const int bh = blockIdx.x;
  const int head = bh & 3;
  if (threadIdx.x < 48) {
    const int c = threadIdx.x;
    const float* src = attnF + ((size_t)bh * 48 + c) * 48;
    const float rqc = rq[bh * 48 + c];
    const float t = temp[head];
    float row[48];
    float mx = -3.0e38f;
    #pragma unroll
    for (int d = 0; d < 48; ++d) {
      float v = src[d] * rqc * rk[bh * 48 + d] * t;
      row[d] = v;
      mx = fmaxf(mx, v);
    }
    float sum = 0.f;
    #pragma unroll
    for (int d = 0; d < 48; ++d) {
      float e = __expf(row[d] - mx);
      row[d] = e;
      sum += e;
    }
    const float inv = 1.f / sum;
    half_t* dst = attnh + ((size_t)bh * 48 + c) * 64;
    #pragma unroll
    for (int d = 0; d < 48; ++d) dst[d] = (half_t)(row[d] * inv);
    #pragma unroll
    for (int d = 48; d < 64; ++d) dst[d] = (half_t)0.f;   // zero K pad
  }
}

// ------------------------- out = attn x v ----------------------------------
// Grid (16 bh, 32 chunk): M=48 (ch), N=512 hw cols/block, K=48 (pad 64).
// attn staged in LDS [48][72]; v 48x16 tiles transposed through per-wave
// LDS [16][72] so B fragments are contiguous ds_load_b128s.
__global__ void __launch_bounds__(256) k_attn_v(const half_t* __restrict__ attnh,
                                                const half_t* __restrict__ vbuf,
                                                half_t* __restrict__ out2h) {
  __shared__ __align__(16) half_t attnA[48 * 72];
  __shared__ __align__(16) half_t vS[8][16 * 72];
  const int bh = blockIdx.x;
  const int head = bh & 3, b = bh >> 2;
  const int lane = threadIdx.x & 31, wave = threadIdx.x >> 5;
  const half_t* vp = vbuf + ((size_t)bh * 48 << 14);

  // stage attn into LDS (row stride 72 to spread banks; cols 64..71 unread)
  for (int idx = threadIdx.x; idx < 48 * 64; idx += 256) {
    int r = idx >> 6, c = idx & 63;
    attnA[r * 72 + c] = attnh[(size_t)bh * 48 * 64 + idx];
  }
  // zero this wave's v-staging area once (pad cols stay zero forever)
  for (int i = lane; i < 16 * 72; i += 32) vS[wave][i] = (half_t)0.f;
  __syncthreads();

  // hoist all 6 A fragments of the 48x64 attn matrix
  v16h aA[3][2];
  #pragma unroll
  for (int mt = 0; mt < 3; ++mt)
    #pragma unroll
    for (int kk = 0; kk < 2; ++kk) {
      int m  = mt * 16 + (lane & 15);
      int kb = kk * 32 + ((lane >> 4) << 4);
      v8h lo = *(const v8h*)&attnA[m * 72 + kb];
      v8h hi = *(const v8h*)&attnA[m * 72 + kb + 8];
      aA[mt][kk] = cat8(lo, hi);
    }

  for (int nt = wave; nt < 32; nt += 8) {
    const int s0 = blockIdx.y * 512 + nt * 16;
    // load v tile [48 ch][16 hw] and transpose into vS[wave] = [16 hw][72 ch]
    #pragma unroll
    for (int i = 0; i < 3; ++i) {
      int d  = i * 16 + (lane & 15);
      int sh = (lane >> 4) << 3;                    // 0 or 8
      v8h row = *(const v8h*)(vp + ((size_t)d << 14) + s0 + sh);
      #pragma unroll
      for (int e = 0; e < 8; ++e) vS[wave][(sh + e) * 72 + d] = row[e];
    }
    asm volatile("s_wait_dscnt 0" ::: "memory");    // order in-wave LDS store->load
    // B fragments from transposed v (K pad 48..63 is zero)
    v16h bV[2];
    #pragma unroll
    for (int kk = 0; kk < 2; ++kk) {
      int kb = kk * 32 + ((lane >> 4) << 4);
      const v8h* p = (const v8h*)&vS[wave][(lane & 15) * 72 + kb];
      bV[kk] = cat8(p[0], p[1]);
    }
    const int hw = s0 + (lane & 15);
    #pragma unroll
    for (int mt = 0; mt < 3; ++mt) {
      v8f acc = vzero8();
      acc = wmma_f16f32(aA[mt][0], bV[0], acc);
      acc = wmma_f16f32(aA[mt][1], bV[1], acc);
      v8h h;
      #pragma unroll
      for (int r = 0; r < 8; ++r) h[r] = (half_t)acc[r];
      size_t o = (size_t)(b * 16384 + hw) * 192 + head * 48 + mt * 16 +
                 ((lane >> 4) << 3);
      *(v8h*)(out2h + o) = h;
    }
    asm volatile("" ::: "memory");                  // keep next tile's stores below reads
  }
}

// ------------------------- output projection GEMM --------------------------
// out[m][n] = sum_k out2h[m][k] * w_proj[k][n];  M=65536, N=192, K=192, f32 out
__global__ void __launch_bounds__(256) k_gemm_proj(const half_t* __restrict__ a,
                                                   const half_t* __restrict__ wT,
                                                   float* __restrict__ out) {
  const int lane = threadIdx.x & 31, wave = threadIdx.x >> 5;
  const int m0 = (blockIdx.x * 8 + wave) << 4;
  const int n0 = blockIdx.y << 6;
  v8f acc[4];
  #pragma unroll
  for (int j = 0; j < 4; ++j) acc[j] = vzero8();
  #pragma unroll
  for (int k0 = 0; k0 < 192; k0 += 32) {
    v16h af = load_a(a, 192, m0, k0, lane);
    #pragma unroll
    for (int j = 0; j < 4; ++j) {
      v16h bf = load_bt(wT, 192, n0 + (j << 4), k0, lane);
      acc[j] = wmma_f16f32(af, bf, acc[j]);
    }
  }
  #pragma unroll
  for (int j = 0; j < 4; ++j) {
    int n = n0 + (j << 4) + (lane & 15);
    #pragma unroll
    for (int r = 0; r < 8; ++r) {
      int m = m0 + ((lane >> 4) << 3) + r;
      out[(size_t)m * 192 + n] = acc[j][r];
    }
  }
}

// ---------------------------------------------------------------------------
extern "C" void kernel_launch(void* const* d_in, const int* in_sizes, int n_in,
                              void* d_out, int out_size, void* d_ws, size_t ws_size,
                              hipStream_t stream) {
  (void)in_sizes; (void)n_in; (void)out_size; (void)ws_size;
  const float* x      = (const float*)d_in[0];   // [4,128,128,192]
  const float* w_qkv  = (const float*)d_in[1];   // [192,576]
  const float* w_dw   = (const float*)d_in[2];   // [3,3,1,576]
  const float* w_proj = (const float*)d_in[3];   // [192,192]
  const float* temp   = (const float*)d_in[4];   // [4]
  float* out = (float*)d_out;

  char* ws = (char*)d_ws;
  half_t* xh    = (half_t*)(ws + OFF_XH);
  half_t* wqT   = (half_t*)(ws + OFF_WQT);
  half_t* wpT   = (half_t*)(ws + OFF_WPT);
  half_t* qkvP  = (half_t*)(ws + OFF_QKVP);
  half_t* qbuf  = (half_t*)(ws + OFF_Q);
  half_t* kbuf  = (half_t*)(ws + OFF_K);
  half_t* vbuf  = (half_t*)(ws + OFF_V);
  float*  rq    = (float*)(ws + OFF_RQ);
  float*  rk    = (float*)(ws + OFF_RK);
  half_t* attnh = (half_t*)(ws + OFF_ATTN);
  float*  attnF = (float*)(ws + OFF_ATTNF);
  half_t* out2h = (half_t*)(ws + OFF_XH);        // alias xh (dead after GEMM1)

  // 1) f32 -> f16 conversions / weight transposes / zero attn accumulator
  k_cvt_x<<<12288, 256, 0, stream>>>(x, xh, 12582912 / 4);
  k_transpose_w<<<(192 * 576 + 255) / 256, 256, 0, stream>>>(w_qkv, wqT, 192, 576);
  k_transpose_w<<<(192 * 192 + 255) / 256, 256, 0, stream>>>(w_proj, wpT, 192, 192);
  k_zero_f32<<<(16 * 48 * 48 + 255) / 256, 256, 0, stream>>>(attnF, 16 * 48 * 48);
  // 2) qkv = x @ w_qkv  (planar D)
  k_gemm_qkv<<<dim3(512, 9), 256, 0, stream>>>(xh, wqT, qkvP);
  // 3) depthwise 3x3 SAME, split into head-planar q/k/v
  k_dwconv<<<dim3(256, 576), 256, 0, stream>>>(qkvP, w_dw, qbuf, kbuf, vbuf);
  // 4) row rsqrt-norms for q then k (contiguous 1536 rows)
  k_rownorm<<<1536, 256, 0, stream>>>(qbuf, rq);
  // 5) attn = q k^T (K split across 8 blocks/bh, global f32 atomic reduce)
  k_attn_partial<<<dim3(16, 8), 256, 0, stream>>>(qbuf, kbuf, attnF);
  // 6) fused normalization scales + temperature + softmax -> f16 [16][48][64]
  k_softmax<<<16, 64, 0, stream>>>(attnF, rq, rk, temp, attnh);
  // 7) out2 = attn @ v  -> f16 [b,hw,192]
  k_attn_v<<<dim3(16, 32), 256, 0, stream>>>(attnh, vbuf, out2h);
  // 8) final projection -> f32 NHWC output
  k_gemm_proj<<<dim3(512, 3), 256, 0, stream>>>(out2h, wpT, out);
}
